// MultiHeadAttention_81716047774274
// MI455X (gfx1250) — matrix operand order repaired and checked by reading, not compiled, not run on hardware
//
#include <hip/hip_runtime.h>
#include <hip/hip_bf16.h>

// ---------------- problem constants ----------------
constexpr int B_  = 4;
constexpr int L_  = 2048;
constexpr int DM_ = 1024;   // d_model
constexpr int H_  = 16;
constexpr int DQ_ = 64;     // d_qkv
// SCALE = sqrt(D_MODEL) = 32  -> scores are divided by 32 before softmax

// ---------------- types ----------------
typedef _Float16 h16;
typedef __attribute__((ext_vector_type(16))) _Float16 hv16;
typedef __attribute__((ext_vector_type(8)))  _Float16 hv8;
typedef __attribute__((ext_vector_type(8)))  float    fv8;
typedef __attribute__((ext_vector_type(4)))  int      i32x4;

// ---------------- workspace layout (element offsets, h16 units) ----------------
constexpr size_t N_XH   = (size_t)B_ * L_ * DM_;          // 8,388,608
constexpr size_t N_WT   = (size_t)3 * H_ * DQ_ * DM_;     // 3,145,728
constexpr size_t N_FCW  = (size_t)DM_ * DM_;              // 1,048,576
constexpr size_t N_HB   = (size_t)B_ * H_ * L_ * DQ_;     // 8,388,608
constexpr size_t OFF_XH  = 0;
constexpr size_t OFF_WT  = OFF_XH + N_XH;
constexpr size_t OFF_FCW = OFF_WT + N_WT;
constexpr size_t OFF_Q   = OFF_FCW + N_FCW;
constexpr size_t OFF_K   = OFF_Q + N_HB;
constexpr size_t OFF_VT  = OFF_K + N_HB;
constexpr size_t OFF_ATT = OFF_VT + N_HB;
// total ~92.3 MB of workspace

// ---------------- gfx1250 async-to-LDS path ----------
#if __has_builtin(__builtin_amdgcn_global_load_async_to_lds_b128)
#define HAVE_ASYNC_LDS 1
#else
#define HAVE_ASYNC_LDS 0
#endif

__device__ __forceinline__ void wait_async0() {
#if __has_builtin(__builtin_amdgcn_s_wait_asynccnt)
  __builtin_amdgcn_s_wait_asynccnt(0);
#else
  asm volatile("s_wait_asynccnt 0" ::: "memory");
#endif
}

// copy 16 bytes global -> LDS (async DMA if available, else load+ds_store)
__device__ __forceinline__ void copy16_g2l(const h16* g, h16* l) {
#if HAVE_ASYNC_LDS
  __builtin_amdgcn_global_load_async_to_lds_b128(
      (__attribute__((address_space(1))) i32x4*)(g),
      (__attribute__((address_space(3))) i32x4*)(l), 0, 0);
#else
  *(hv8*)l = *(const hv8*)g;
#endif
}

// ---------------- WMMA fragment loaders (ISA 7.12.2 layouts, wave32) ----------------
// A 16x32 f16: lanes 0-15 -> M=lane, K = 0..7 (v0-3) and 16..23 (v4-7)
//              lanes 16-31 -> M=lane-16, K = 8..15 and 24..31
__device__ __forceinline__ hv16 load_a_frag(const h16* tile, int ld) {
  const int lane = threadIdx.x & 31;
  const int row  = lane & 15;
  const int kb   = (lane >> 4) << 3;          // 0 or 8
  const h16* p = tile + (size_t)row * ld + kb;
  hv8 lo = *(const hv8*)(p);                  // K = kb .. kb+7
  hv8 hi = *(const hv8*)(p + 16);             // K = kb+16 .. kb+23
  hv16 r;
#pragma unroll
  for (int i = 0; i < 8; ++i) { r[i] = lo[i]; r[i + 8] = hi[i]; }
  return r;
}

// B 32x16 f16 from "transposed" storage: bt[n*ld + k] = B[k][n].
// Per lane K = kb..kb+7 and kb+16..kb+23 (kb = 0 or 8) -> two 16B loads.
__device__ __forceinline__ hv16 load_bT_frag(const h16* bt, int ld) {
  const int lane = threadIdx.x & 31;
  const int n    = lane & 15;
  const int kb   = (lane >> 4) << 3;          // 0 or 8
  const h16* p = bt + (size_t)n * ld + kb;
  hv8 lo = *(const hv8*)(p);
  hv8 hi = *(const hv8*)(p + 16);
  hv16 r;
#pragma unroll
  for (int i = 0; i < 8; ++i) { r[i] = lo[i]; r[i + 8] = hi[i]; }
  return r;
}

#define WMMA_F32_F16(a, b, c) \
  __builtin_amdgcn_wmma_f32_16x16x32_f16(false, (a), false, (b), (short)0, (c), false, false)

// ---------------- stage 0: f32 -> f16 conversions ----------------
__global__ void mha_cvt_x(const float* __restrict__ x, h16* __restrict__ xh, int n) {
  int i = blockIdx.x * blockDim.x + threadIdx.x;
  if (i < n) xh[i] = (h16)x[i];
}

// wT[((which*H + h)*DQ + d)*DM + m] = w_which[h][m][d]
__global__ void mha_cvt_w(const float* __restrict__ wq, const float* __restrict__ wk,
                          const float* __restrict__ wv, h16* __restrict__ wT) {
  int i = blockIdx.x * blockDim.x + threadIdx.x;   // i indexes the OUTPUT
  int m = i % DM_;
  int d = (i / DM_) % DQ_;
  int h = (i / (DM_ * DQ_)) % H_;
  int which = i / (DM_ * DQ_ * H_);
  const float* src = (which == 0) ? wq : (which == 1) ? wk : wv;
  wT[i] = (h16)src[(size_t)h * DM_ * DQ_ + (size_t)m * DQ_ + d];
}

__global__ void mha_cvt_fcw(const float* __restrict__ fcw, h16* __restrict__ fcwh, int n) {
  int i = blockIdx.x * blockDim.x + threadIdx.x;
  if (i < n) fcwh[i] = (h16)fcw[i];
}

// ---------------- shared helper: stage a 64(n) x 32(k) weight chunk to LDS ----
// Wsrc is row-major [n][DM_]; chunk columns k0..k0+31 are contiguous per row.
// 128 threads: 2 threads x 32B cover one row.
__device__ __forceinline__ void stage_b_chunk(const h16* __restrict__ Wsrc,
                                              int k0, h16* b_l) {
  const int t = threadIdx.x;       // 0..127
  const int row = t >> 1;          // 0..63
  const int off = (t & 1) * 16;    // 0 or 16
  const h16* src = Wsrc + (size_t)row * DM_ + k0 + off;
  h16* dst = b_l + row * 32 + off;
  copy16_g2l(src, dst);
  copy16_g2l(src + 8, dst + 8);
}

// ---------------- stage 1: QKV projection ----------------
// grid: (L/128, B*H, 3), block 128 (4 waves). Wave computes a 32x64 tile.
// Weight chunk (shared by all 4 waves) is staged into LDS via async DMA,
// double-buffered; A fragments are register-prefetched.
__global__ __launch_bounds__(128, 1)
void mha_qkv(const h16* __restrict__ xh, const h16* __restrict__ wT,
             h16* __restrict__ qbuf, h16* __restrict__ kbuf,
             h16* __restrict__ vT) {
  __shared__ __align__(16) h16 b_lds[2][DQ_ * 32];   // 2 x 4KB, [n=64][k=32]

  const int wave = threadIdx.x >> 5;
  const int lane = threadIdx.x & 31;
  const int which = blockIdx.z;               // 0=q 1=k 2=v
  const int bh = blockIdx.y;
  const int b = bh >> 4;
  const int row0 = blockIdx.x * 128 + wave * 32;

  const h16* A  = xh + ((size_t)b * L_ + row0) * DM_;
  const h16* Wt = wT + (size_t)(which * H_ + (bh & 15)) * DQ_ * DM_;  // [d][m]

  fv8 acc[2][4];
#pragma unroll
  for (int rt = 0; rt < 2; ++rt)
#pragma unroll
    for (int t = 0; t < 4; ++t) acc[rt][t] = (fv8){};

  stage_b_chunk(Wt, 0, b_lds[0]);                 // prefetch B chunk 0
  hv16 a_cur0 = load_a_frag(A, DM_);              // rows row0..+15
  hv16 a_cur1 = load_a_frag(A + (size_t)16 * DM_, DM_);  // rows row0+16..+31

  constexpr int NKC = DM_ / 32;                   // 32 K-chunks
  for (int c = 0; c < NKC; ++c) {
    wait_async0();
    __syncthreads();
    if (c + 1 < NKC) stage_b_chunk(Wt, (c + 1) * 32, b_lds[(c + 1) & 1]);
    hv16 a_n0 = a_cur0, a_n1 = a_cur1;
    if (c + 1 < NKC) {
      a_n0 = load_a_frag(A + (c + 1) * 32, DM_);
      a_n1 = load_a_frag(A + (size_t)16 * DM_ + (c + 1) * 32, DM_);
    }
    const h16* bl = b_lds[c & 1];
#pragma unroll
    for (int t = 0; t < 4; ++t) {
      hv16 bf = load_bT_frag(bl + t * 16 * 32, 32);
      acc[0][t] = WMMA_F32_F16(a_cur0, bf, acc[0][t]);
      acc[1][t] = WMMA_F32_F16(a_cur1, bf, acc[1][t]);
    }
    a_cur0 = a_n0;
    a_cur1 = a_n1;
  }

  const int col = lane & 15;
  const int rbase = (lane >> 4) << 3;
#pragma unroll
  for (int rt = 0; rt < 2; ++rt) {
#pragma unroll
    for (int t = 0; t < 4; ++t) {
#pragma unroll
      for (int r = 0; r < 8; ++r) {
        h16 val = (h16)acc[rt][t][r];
        int l = row0 + rt * 16 + rbase + r;
        int d = t * 16 + col;
        if (which == 0)      qbuf[((size_t)bh * L_ + l) * DQ_ + d] = val;
        else if (which == 1) kbuf[((size_t)bh * L_ + l) * DQ_ + d] = val;
        else                 vT[((size_t)bh * DQ_ + d) * L_ + l] = val;  // transposed
      }
    }
  }
}

// ---------------- stage 2: flash attention ----------------
// grid: (L/128, B*H), block 256 (8 waves, 128 query rows).
// Each 32-key K/V chunk is staged into LDS ONCE per block (async DMA,
// double-buffered); 8 waves consume it. 8 WMMAs per chunk per wave.
constexpr int KCH = 32;                    // keys per chunk
constexpr int NCH = L_ / KCH;              // 64 chunks

// 256 threads cooperatively stage K chunk (32x64 h16) and V chunk (64x32 h16)
__device__ __forceinline__ void stage_kv_chunk(const h16* __restrict__ Kg,
                                               const h16* __restrict__ Vg,
                                               int key0, h16* k_l, h16* v_l) {
  const int t = threadIdx.x;               // 0..255
  {                                        // K: row=key (32), 64 halves/row
    const int row = t >> 3;                // 8 threads x 16B cover a row
    const int off = (t & 7) * 8;
    copy16_g2l(Kg + (size_t)(key0 + row) * DQ_ + off, k_l + row * 64 + off);
  }
  {                                        // V^T: row=d (64), 32 halves/row
    const int row = t >> 2;                // 4 threads x 16B cover a row
    const int off = (t & 3) * 8;
    copy16_g2l(Vg + (size_t)row * L_ + key0 + off, v_l + row * 32 + off);
  }
}

__global__ __launch_bounds__(256, 1)
void mha_attn(const h16* __restrict__ qbuf, const h16* __restrict__ kbuf,
              const h16* __restrict__ vT, const int* __restrict__ mask,
              h16* __restrict__ attn) {
  __shared__ __align__(16) h16 k_lds[2][KCH * DQ_];   // 2 x 4KB
  __shared__ __align__(16) h16 v_lds[2][DQ_ * KCH];   // 2 x 4KB
  __shared__ __align__(16) h16 p_lds[8][16 * 32];     // per-wave P transpose

  const int wave = threadIdx.x >> 5;
  const int lane = threadIdx.x & 31;
  const int bh = blockIdx.y;
  const int b = bh >> 4;
  const int h = bh & 15;
  const int q0 = blockIdx.x * 128 + wave * 16;

  const h16* Q  = qbuf + ((size_t)bh * L_ + q0) * DQ_;
  const h16* Kg = kbuf + (size_t)bh * L_ * DQ_;
  const h16* Vg = vT + (size_t)bh * DQ_ * L_;
  const int* mrow_ptr = mask + b * L_;

  hv16 aq0 = load_a_frag(Q, DQ_);        // d = 0..31 (loop invariant)
  hv16 aq1 = load_a_frag(Q + 32, DQ_);   // d = 32..63

  float mrow[8], lrow[8], alpha[8];
  fv8 acc[4];
#pragma unroll
  for (int r = 0; r < 8; ++r) { mrow[r] = -1e30f; lrow[r] = 0.f; }
#pragma unroll
  for (int t = 0; t < 4; ++t) acc[t] = (fv8){};

  const float inv_scale = 1.0f / 32.0f;  // 1/sqrt(D_MODEL)
  const int col = lane & 15;
  const int rbase = (lane >> 4) << 3;
  h16* pl = p_lds[wave];

  stage_kv_chunk(Kg, Vg, 0, k_lds[0], v_lds[0]);   // prefetch chunk 0

  for (int c = 0; c < NCH; ++c) {
    wait_async0();                  // my chunk-c stage requests are done
    __syncthreads();                // everyone's are done; buf (c-1) free
    if (c + 1 < NCH)                // overlap chunk c+1 DMA with chunk c math
      stage_kv_chunk(Kg, Vg, (c + 1) * KCH, k_lds[(c + 1) & 1], v_lds[(c + 1) & 1]);

    const h16* kc = k_lds[c & 1];   // [key][d], ld = 64
    const h16* vc = v_lds[c & 1];   // [d][key], ld = 32
    const int key0 = c * KCH;

    // S[16x32] = Q(16x64) * K^T(64x32): two 16x16 N-tiles, 2 k-steps each
    fv8 s[2];
#pragma unroll
    for (int t = 0; t < 2; ++t) {
      const h16* kt = kc + (size_t)(t * 16) * DQ_;
      fv8 cc = (fv8){};
      cc = WMMA_F32_F16(aq0, load_bT_frag(kt, DQ_), cc);
      cc = WMMA_F32_F16(aq1, load_bT_frag(kt + 32, DQ_), cc);
      s[t] = cc;
    }
    // scale + mask (mask indexed by key column = this lane's N)
#pragma unroll
    for (int t = 0; t < 2; ++t) {
      int mv = mrow_ptr[key0 + t * 16 + col];
#pragma unroll
      for (int r = 0; r < 8; ++r) {
        float v = s[t][r] * inv_scale;
        s[t][r] = (mv == 0) ? -3.125e7f : v;   // -1e9 / 32
      }
    }
    // online softmax; a row's 16 columns live in one 16-lane half
#pragma unroll
    for (int r = 0; r < 8; ++r) {
      float v = fmaxf(s[0][r], s[1][r]);
#pragma unroll
      for (int off = 1; off < 16; off <<= 1) v = fmaxf(v, __shfl_xor(v, off, 32));
      float mn = fmaxf(mrow[r], v);
      alpha[r] = __expf(mrow[r] - mn);
      mrow[r] = mn;
      float p0 = __expf(s[0][r] - mn);
      float p1 = __expf(s[1][r] - mn);
      s[0][r] = p0; s[1][r] = p1;
      float sum = p0 + p1;
#pragma unroll
      for (int off = 1; off < 16; off <<= 1) sum += __shfl_xor(sum, off, 32);
      lrow[r] = lrow[r] * alpha[r] + sum;
    }
#pragma unroll
    for (int t = 0; t < 4; ++t)
#pragma unroll
      for (int r = 0; r < 8; ++r) acc[t][r] *= alpha[r];

    // transpose P (C layout) -> A layout via per-wave LDS tile
#pragma unroll
    for (int r = 0; r < 8; ++r) {
      pl[(rbase + r) * 32 + col]      = (h16)s[0][r];
      pl[(rbase + r) * 32 + 16 + col] = (h16)s[1][r];
    }
    asm volatile("s_wait_dscnt 0" ::: "memory");   // wave-internal cross-lane RAW
    hv16 pa = load_a_frag(pl, 32);

    // acc += P(16x32) * V(32x64); v_lds is [d][key] -> contiguous-K B frags
#pragma unroll
    for (int t = 0; t < 4; ++t)
      acc[t] = WMMA_F32_F16(pa, load_bT_frag(vc + (size_t)(t * 16) * KCH, KCH), acc[t]);
  }

  // normalize rows and store f16 to [b][l][h*64+d]
#pragma unroll
  for (int r = 0; r < 8; ++r) {
    float invl = 1.0f / lrow[r];
#pragma unroll
    for (int t = 0; t < 4; ++t) {
      float o = acc[t][r] * invl;
      attn[(size_t)(b * L_ + q0 + rbase + r) * (H_ * DQ_) + h * DQ_ + t * 16 + col] = (h16)o;
    }
  }
}

// ---------------- stage 3: FC projection + bias (fp32 out) ----------------
// grid: (B*L/128, DM/64), block 128 (4 waves). Wave computes a 32x64 tile.
// fc_w chunk staged into LDS (async, double-buffered), shared by all waves.
__global__ __launch_bounds__(128, 1)
void mha_fc(const h16* __restrict__ attn, const h16* __restrict__ fcwh,
            const float* __restrict__ bias, float* __restrict__ out) {
  __shared__ __align__(16) h16 b_lds[2][DQ_ * 32];   // 2 x 4KB, [n=64][k=32]

  const int wave = threadIdx.x >> 5;
  const int lane = threadIdx.x & 31;
  const int row0 = blockIdx.x * 128 + wave * 32;
  const int col0 = blockIdx.y * 64;

  const h16* A = attn + (size_t)row0 * DM_;
  const h16* W = fcwh + (size_t)col0 * DM_;        // [o][i], contiguous K

  fv8 acc[2][4];
#pragma unroll
  for (int rt = 0; rt < 2; ++rt)
#pragma unroll
    for (int t = 0; t < 4; ++t) acc[rt][t] = (fv8){};

  stage_b_chunk(W, 0, b_lds[0]);
  hv16 a_cur0 = load_a_frag(A, DM_);
  hv16 a_cur1 = load_a_frag(A + (size_t)16 * DM_, DM_);

  constexpr int NKC = DM_ / 32;
  for (int c = 0; c < NKC; ++c) {
    wait_async0();
    __syncthreads();
    if (c + 1 < NKC) stage_b_chunk(W, (c + 1) * 32, b_lds[(c + 1) & 1]);
    hv16 a_n0 = a_cur0, a_n1 = a_cur1;
    if (c + 1 < NKC) {
      a_n0 = load_a_frag(A + (c + 1) * 32, DM_);
      a_n1 = load_a_frag(A + (size_t)16 * DM_ + (c + 1) * 32, DM_);
    }
    const h16* bl = b_lds[c & 1];
#pragma unroll
    for (int t = 0; t < 4; ++t) {
      hv16 bf = load_bT_frag(bl + t * 16 * 32, 32);
      acc[0][t] = WMMA_F32_F16(a_cur0, bf, acc[0][t]);
      acc[1][t] = WMMA_F32_F16(a_cur1, bf, acc[1][t]);
    }
    a_cur0 = a_n0;
    a_cur1 = a_n1;
  }

  const int col = lane & 15;
  const int rbase = (lane >> 4) << 3;
#pragma unroll
  for (int rt = 0; rt < 2; ++rt) {
#pragma unroll
    for (int t = 0; t < 4; ++t) {
#pragma unroll
      for (int r = 0; r < 8; ++r) {
        int o = col0 + t * 16 + col;
        out[(size_t)(row0 + rt * 16 + rbase + r) * DM_ + o] = acc[rt][t][r] + bias[o];
      }
    }
  }
}

// ---------------- launch ----------------
extern "C" void kernel_launch(void* const* d_in, const int* in_sizes, int n_in,
                              void* d_out, int out_size, void* d_ws, size_t ws_size,
                              hipStream_t stream) {
  const float* x    = (const float*)d_in[0];
  const int*   mask = (const int*)d_in[1];
  const float* wq   = (const float*)d_in[2];
  const float* wk   = (const float*)d_in[3];
  const float* wv   = (const float*)d_in[4];
  const float* fcw  = (const float*)d_in[5];
  const float* fcb  = (const float*)d_in[6];
  float* out = (float*)d_out;

  h16* ws   = (h16*)d_ws;
  h16* xh   = ws + OFF_XH;
  h16* wT   = ws + OFF_WT;
  h16* fcwh = ws + OFF_FCW;
  h16* qbuf = ws + OFF_Q;
  h16* kbuf = ws + OFF_K;
  h16* vTb  = ws + OFF_VT;
  h16* attn = ws + OFF_ATT;

  // stage 0: conversions
  mha_cvt_x<<<(int)(N_XH / 256), 256, 0, stream>>>(x, xh, (int)N_XH);
  mha_cvt_w<<<(int)(N_WT / 256), 256, 0, stream>>>(wq, wk, wv, wT);
  mha_cvt_fcw<<<(int)(N_FCW / 256), 256, 0, stream>>>(fcw, fcwh, (int)N_FCW);

  // stage 1: QKV projections (LDS-staged weights)
  mha_qkv<<<dim3(L_ / 128, B_ * H_, 3), 128, 0, stream>>>(xh, wT, qbuf, kbuf, vTb);

  // stage 2: flash attention (async LDS staging, double-buffered)
  mha_attn<<<dim3(L_ / 128, B_ * H_), 256, 0, stream>>>(qbuf, kbuf, vTb, mask, attn);

  // stage 3: FC + bias (LDS-staged weights)
  mha_fc<<<dim3((B_ * L_) / 128, DM_ / 64), 128, 0, stream>>>(attn, fcwh, fcb, out);
}
